// AffinityLayer_16801912062283
// MI455X (gfx1250) — matrix operand order repaired and checked by reading, not compiled, not run on hardware
//
#include <hip/hip_runtime.h>
#include <hip/hip_bf16.h>
#include <math.h>

// ---------------------------------------------------------------------------
// AffinityLayer recurrent scan for MI455X (gfx1250, wave32, WMMA).
//
// Shapes: b=128, n=512, x_len=512, h_len=512, fan_in=1024.
//   * 8 persistent workgroups, each owns a 16-row batch tile (independent
//     recurrences -> scan loop lives in the kernel, no inter-WG sync).
//   * Weights converted once/launch to bf16, pre-swizzled into per-lane-
//     contiguous v_wmma B-fragment order in d_ws (2 MB, L2-resident) so each
//     fragment is exactly 2x global_load_b128 per lane.
//   * Per step each of the 8 waves computes [16 x 1024] x [1024 x 128] with
//     256 v_wmma_f32_16x16x32_bf16: its 64 'a' columns and the matching 64
//     gate columns, so the sigmoid/tanh blend happens in-register.
//   * K-tile loop unrolled 2x: at ~2 waves/SIMD occupancy the wave's own
//     instruction window is the only L2-latency buffer, so widen it.
//   * Branch-free sigmoid/tanh (v_exp + v_rcp), no EXEC-divergent libcalls.
//   * LayerNorm: shfl_xor butterfly inside 16-lane halves (C-fragment rows),
//     then LDS ds_add_f32 atomics across waves.
// ---------------------------------------------------------------------------

typedef __attribute__((ext_vector_type(16))) __bf16 v16bf;
typedef __attribute__((ext_vector_type(8)))  float v8f;
typedef __attribute__((ext_vector_type(4)))  unsigned int v4u;

#define B_TOT   128
#define N_T     512
#define XLEN    512
#define HLEN    512
#define FANIN   1024   // h + x
#define NOUT    1024   // a columns (512) + gate columns (512)
#define MTILE   16
#define KTILES  (FANIN / 32)   // 32
#define NTILES  (NOUT / 16)    // 64
#define CAT_STRIDE (FANIN + 8) // bf16 elems; pad -> 4-dword bank skew per row
#define EPS     1e-5f

union FragBF {
    v16bf        v;
    v4u          q[2];
    unsigned int u[8];
};

static __device__ __forceinline__ unsigned short f32_to_bf16(float f) {
    unsigned int u = __float_as_uint(f);
    unsigned int r = u + 0x7FFFu + ((u >> 16) & 1u);   // round-to-nearest-even
    return (unsigned short)(r >> 16);
}

// ---------------------------------------------------------------------------
// Prep: W_a[512,1024], W_g[512,1024] (fp32, row-major over fan_in) ->
// bf16 B-fragments for v_wmma_f32_16x16x32_bf16.
// B[k][n] = (n < 512) ? W_a[n][k] : W_g[n-512][k]
// Fragment layout used here (16-bit B, contiguous k, same as A): lane = n%16 (+16 for
// K-half 16..31), VGPR v holds packed {K=2v (lo16), K=2v+1 (hi16)}.
// Memory layout: wq[(((ntile*KTILES + kt)*32 + lane)*8 + v]  -- each lane's
// 8 dwords contiguous -> fragment loads are 2x b128 per lane, wave covers the
// whole 1 KB fragment contiguously.
// ---------------------------------------------------------------------------
__global__ void prep_weights_kernel(const float* __restrict__ Wa,
                                    const float* __restrict__ Wg,
                                    unsigned int* __restrict__ wq) {
    const int blk   = blockIdx.x;          // 0 .. NTILES*KTILES-1
    const int ntile = blk >> 5;            // /32
    const int kt    = blk & 31;
    const int t     = threadIdx.x;         // 0..255
    const int v     = t >> 5;
    const int lane  = t & 31;
    const int ncol  = lane & 15;
    const int half  = lane >> 4;

    const int n = ntile * 16 + ncol;
    const int k = kt * 32 + 2 * v + half * 16;

    const float* src = (n < HLEN) ? (Wa + (size_t)n * FANIN)
                                  : (Wg + (size_t)(n - HLEN) * FANIN);
    unsigned int lo = f32_to_bf16(src[k]);
    unsigned int hi = f32_to_bf16(src[k + 1]);
    wq[(((size_t)(ntile * KTILES + kt)) * 32 + lane) * 8 + v] = lo | (hi << 16);
}

// ---------------------------------------------------------------------------
// Main persistent scan kernel: grid = 8 blocks x 256 threads (8 waves).
// ---------------------------------------------------------------------------
__global__ __launch_bounds__(256) void
affinity_scan_kernel(const float* __restrict__ X,
                     const unsigned int* __restrict__ wq,
                     const float* __restrict__ b_a,
                     const float* __restrict__ b_g,
                     const float* __restrict__ gamma,
                     const float* __restrict__ beta,
                     float* __restrict__ out) {
    __shared__ unsigned short s_cat[MTILE * CAT_STRIDE];  // [16][1032] bf16
    __shared__ float s_red[MTILE * 2];                    // per-row sum, sumsq

    const int tid    = threadIdx.x;
    const int wave   = tid >> 5;        // 0..7
    const int lane   = tid & 31;
    const int ncol   = lane & 15;
    const int half   = lane >> 4;       // C-frag: row = r + half*8
    const int wgBase = blockIdx.x * MTILE;

    // Per-lane column metadata (time-invariant): this wave's 64 h-columns.
    float bA[4], bG[4], gm[4], bt[4];
    int   nh[4];
#pragma unroll
    for (int j = 0; j < 4; ++j) {
        nh[j] = wave * 64 + j * 16 + ncol;          // h-index 0..511
        bA[j] = b_a[nh[j]];
        bG[j] = b_g[nh[j]];
        gm[j] = gamma[nh[j]];
        bt[j] = beta[nh[j]];
    }

    // ---- init cat: h part = 0 (h0), x part = X[:, 0, :] ----
    for (int idx = tid; idx < MTILE * HLEN; idx += 256) {
        const int m = idx >> 9, c = idx & 511;
        s_cat[m * CAT_STRIDE + c] = 0;
        s_cat[m * CAT_STRIDE + HLEN + c] =
            f32_to_bf16(X[((size_t)(wgBase + m) * N_T + 0) * XLEN + c]);
    }
    __syncthreads();

    const v8f zacc = {0.f, 0.f, 0.f, 0.f, 0.f, 0.f, 0.f, 0.f};

    for (int t = 0; t < N_T; ++t) {
        // ================= GEMM: [16 x 1024] x [1024 x 128] per wave ========
        v8f accA[4], accG[4];
#pragma unroll
        for (int j = 0; j < 4; ++j) { accA[j] = zacc; accG[j] = zacc; }

        const int m_a = lane & 15;   // A-frag row
#pragma unroll 2
        for (int kt = 0; kt < KTILES; ++kt) {
            // A fragment (16-bit A 16x32 layout, matching B): VGPR v -> K=2v;
            // lanes 16..31 read K+16.  These 8 dwords form two
            // contiguous 16B runs -> compiler emits 2x ds_load_b128.
            FragBF af;
#pragma unroll
            for (int v = 0; v < 8; ++v) {
                const int k = 2 * v + half * 16;
                af.u[v] = *(const unsigned int*)&s_cat[m_a * CAT_STRIDE + kt * 32 + k];
            }
            // 'a' columns: ntile = wave*4 + j
#pragma unroll
            for (int j = 0; j < 4; ++j) {
                const v4u* wp = (const v4u*)(wq +
                    (((size_t)((wave * 4 + j) * KTILES + kt)) * 32 + lane) * 8);
                FragBF bf_;
                bf_.q[0] = wp[0];
                bf_.q[1] = wp[1];
                accA[j] = __builtin_amdgcn_wmma_f32_16x16x32_bf16(
                    false, af.v, false, bf_.v, (short)0, accA[j], false, false);
            }
            // gate columns: ntile = 32 + wave*4 + j
#pragma unroll
            for (int j = 0; j < 4; ++j) {
                const v4u* wp = (const v4u*)(wq +
                    (((size_t)((32 + wave * 4 + j) * KTILES + kt)) * 32 + lane) * 8);
                FragBF bf_;
                bf_.q[0] = wp[0];
                bf_.q[1] = wp[1];
                accG[j] = __builtin_amdgcn_wmma_f32_16x16x32_bf16(
                    false, af.v, false, bf_.v, (short)0, accG[j], false, false);
            }
        }

        // ================= blend + per-row partial sums =====================
        // Branch-free activations (no EXEC-divergent tanhf libcall):
        //   sigmoid(g) = 1 / (1 + e^-g)
        //   tanh(a)    = 1 - 2 / (1 + e^{2a})   (saturates correctly at +-inf)
        float ht[4][8];
        float lsum[8], lsq[8];
#pragma unroll
        for (int r = 0; r < 8; ++r) { lsum[r] = 0.f; lsq[r] = 0.f; }
#pragma unroll
        for (int j = 0; j < 4; ++j) {
#pragma unroll
            for (int r = 0; r < 8; ++r) {
                const float a_raw = accA[j][r] + bA[j];
                const float g_raw = accG[j][r] + bG[j];
                const float al    = __frcp_rn(1.0f + __expf(-g_raw));
                const float th    = 1.0f - 2.0f * __frcp_rn(1.0f + __expf(2.0f * a_raw));
                const float v     = al * th + (1.0f - al) * a_raw;
                ht[j][r] = v;
                lsum[r] += v;
                lsq[r]  += v * v;
            }
        }
        // butterfly over the 16 lanes of this half (N = 0..15)
#pragma unroll
        for (int r = 0; r < 8; ++r) {
#pragma unroll
            for (int m = 1; m < 16; m <<= 1) {
                lsum[r] += __shfl_xor(lsum[r], m, 16);
                lsq[r]  += __shfl_xor(lsq[r],  m, 16);
            }
        }

        __syncthreads();                       // all A-frag LDS reads done
        if (tid < MTILE * 2) s_red[tid] = 0.f;
        __syncthreads();

        if (ncol == 0) {                        // 2 lanes/wave carry row sums
#pragma unroll
            for (int r = 0; r < 8; ++r) {
                const int M = r + half * 8;
                atomicAdd(&s_red[M * 2 + 0], lsum[r]);
                atomicAdd(&s_red[M * 2 + 1], lsq[r]);
            }
        }
        // overlap: stage next x_t into LDS while the reduction lands
        if (t + 1 < N_T) {
            for (int idx = tid; idx < MTILE * XLEN; idx += 256) {
                const int m = idx >> 9, c = idx & 511;
                s_cat[m * CAT_STRIDE + HLEN + c] =
                    f32_to_bf16(X[((size_t)(wgBase + m) * N_T + (t + 1)) * XLEN + c]);
            }
        }
        __syncthreads();

        // ================= LayerNorm + writeback ============================
#pragma unroll
        for (int r = 0; r < 8; ++r) {
            const int   M   = r + half * 8;
            const float mu  = s_red[M * 2 + 0] * (1.0f / (float)HLEN);
            const float var = s_red[M * 2 + 1] * (1.0f / (float)HLEN) - mu * mu;
            const float inv = rsqrtf(var + EPS);
#pragma unroll
            for (int j = 0; j < 4; ++j) {
                const float hn = (ht[j][r] - mu) * inv * gm[j] + bt[j];
                out[((size_t)(wgBase + M) * N_T + t) * HLEN + nh[j]] = hn;
                s_cat[M * CAT_STRIDE + nh[j]] = f32_to_bf16(hn);  // next h
            }
        }
        __syncthreads();   // cat fully rebuilt for step t+1
    }
}

// ---------------------------------------------------------------------------
extern "C" void kernel_launch(void* const* d_in, const int* in_sizes, int n_in,
                              void* d_out, int out_size, void* d_ws, size_t ws_size,
                              hipStream_t stream) {
    const float* X     = (const float*)d_in[0];  // [128,512,512]
    const float* W_a   = (const float*)d_in[1];  // [512,1024]
    const float* W_g   = (const float*)d_in[2];  // [512,1024]
    const float* b_a   = (const float*)d_in[3];  // [512]
    const float* b_g   = (const float*)d_in[4];  // [512]
    const float* gamma = (const float*)d_in[5];  // [512]
    const float* beta  = (const float*)d_in[6];  // [512]
    float*       out   = (float*)d_out;          // [128,512,512]

    unsigned int* wq = (unsigned int*)d_ws;      // 2 MB of swizzled bf16 weights

    // One block per (ntile, ktile) = 64*32 blocks, 256 threads (v, lane).
    prep_weights_kernel<<<NTILES * KTILES, 256, 0, stream>>>(W_a, W_g, wq);

    // 8 persistent workgroups, one 16-row batch tile each.
    affinity_scan_kernel<<<B_TOT / MTILE, 256, 0, stream>>>(
        X, wq, b_a, b_g, gamma, beta, out);
}
